// Model_39676907883697
// MI455X (gfx1250) — compile-verified, matrix operand order checked by reading, not run on hardware
//
#include <hip/hip_runtime.h>
#include <hip/hip_bf16.h>

// ---------- CDNA5 WMMA types ----------
typedef __attribute__((ext_vector_type(16))) __bf16 v16bf;
typedef __attribute__((ext_vector_type(8)))  __bf16 v8bf;
typedef __attribute__((ext_vector_type(8)))  float  v8f;

#define WMMA_BF16(a, b, c) \
  __builtin_amdgcn_wmma_f32_16x16x32_bf16(false, (a), false, (b), (short)0, (c), false, false)

// Load a 16x32 bf16 fragment (A-matrix layout; identical per-lane pattern serves as the
// B-fragment of the transposed matrix): lane L<16 holds row L, k in {0..7, 16..23};
// lane L>=16 holds row L-16, k in {8..15, 24..31}.
__device__ __forceinline__ v16bf load_frag(const __bf16* __restrict__ base, int ld) {
  const int lane = threadIdx.x & 31;
  const __bf16* p = base + (size_t)(lane & 15) * (size_t)ld + ((lane >> 4) << 3);
  v8bf lo = *(const v8bf*)(p);
  v8bf hi = *(const v8bf*)(p + 16);
  return __builtin_shufflevector(lo, hi, 0, 1, 2, 3, 4, 5, 6, 7, 8, 9, 10, 11, 12, 13, 14, 15);
}

// ---------- conversion kernels ----------
__global__ __launch_bounds__(256, 1) void cvt_kernel(const float* __restrict__ src,
                                                     __bf16* __restrict__ dst, long long n) {
  long long i = (long long)blockIdx.x * blockDim.x + threadIdx.x;
  if (i < n) dst[i] = (__bf16)src[i];
}

// W [K,N] row-major f32 -> Wt [N,K] row-major bf16
__global__ __launch_bounds__(256, 1) void cvt_transpose_kernel(const float* __restrict__ W,
                                                               __bf16* __restrict__ Wt,
                                                               int K, int N) {
  long long total = (long long)K * N;
  long long i = (long long)blockIdx.x * blockDim.x + threadIdx.x;
  if (i < total) {
    int k = (int)(i / N);
    int n = (int)(i % N);
    Wt[(long long)n * K + k] = (__bf16)W[i];
  }
}

// ---------- projection GEMM ----------
// C[M,384] = A[M,K] @ W(cols c0..c0+383) + bias, epilogue:
//   EPI=0: plain (K-proj)            -> out row-major [M,384] bf16
//   EPI=1: q: (c+bias)*0.125 + mask  -> out row-major [M,384] bf16
//   EPI=2: v: plain, TRANSPOSED out  -> out [B,384,ntok] bf16
template <int EPI>
__global__ __launch_bounds__(256, 1)
void proj_gemm(const __bf16* __restrict__ A, const __bf16* __restrict__ Wt,
               const float* __restrict__ bias, const float* __restrict__ mask,
               __bf16* __restrict__ out, int K, int ntok, int c0) {
  const int lane = threadIdx.x & 31;
  const int wave = threadIdx.x >> 5;
  const int m0 = blockIdx.x * 128 + wave * 16;
  const int n0 = blockIdx.y * 64;

  v8f acc[4] = {};
  const __bf16* Abase = A + (size_t)m0 * K;
  for (int kc = 0; kc < K; kc += 32) {
    v16bf a = load_frag(Abase + kc, K);
#pragma unroll
    for (int t = 0; t < 4; ++t) {
      v16bf b = load_frag(Wt + (size_t)(c0 + n0 + t * 16) * K + kc, K);
      acc[t] = WMMA_BF16(a, b, acc[t]);
    }
  }

  const int cl = lane & 15;
  const int rhalf = (lane >> 4) << 3;
  const int bb = m0 / ntok;            // batch (uniform per wave-tile)
  const int tokbase = (m0 % ntok) + rhalf;
#pragma unroll
  for (int t = 0; t < 4; ++t) {
    const int col = n0 + t * 16 + cl;  // 0..383 (local to the 6-head slice)
    const float bv = bias[c0 + col];
    float mk = 0.0f;
    if (EPI == 1) mk = mask[bb * 64 + (col & 63)];
#pragma unroll
    for (int i = 0; i < 8; ++i) {
      float v = acc[t][i] + bv;
      if (EPI == 1) v = v * 0.125f + mk;  // 64^-0.5 = 0.125
      const int tok = tokbase + i;
      if (EPI == 2) {
        out[((size_t)bb * 384 + col) * ntok + tok] = (__bf16)v;
      } else {
        out[((size_t)bb * ntok + tok) * 384 + col] = (__bf16)v;
      }
    }
  }
}

// ---------- flash attention (transposed-score formulation) ----------
// Q,K: [B, ntok, 384] bf16 (Q pre-scaled + mask-fused); Vt: [B, 384, ntok] bf16.
// O: [B, ntok, 384] bf16. grid = (ntok/64, 6 heads, 8 batch), block = 128 (4 waves).
// Scores computed as S^T = K_block @ Q^T so each lane holds 16 key-scores of ONE query
// (query = lane&15; the other 16 keys live in lane^16) -> softmax reductions are in-lane
// VALU + a single shfl_xor(16), instead of 8x4 ds_bpermute trees per tile.
__global__ __launch_bounds__(128, 1)
void attn_kernel(const __bf16* __restrict__ Q, const __bf16* __restrict__ Kmat,
                 const __bf16* __restrict__ Vt, __bf16* __restrict__ O, int ntok) {
  const int lane = threadIdx.x & 31;
  const int wave = threadIdx.x >> 5;
  const int m0 = blockIdx.x * 64 + wave * 16;
  const int h = blockIdx.y;
  const int b = blockIdx.z;
  const int cl = lane & 15;
  const int half = lane >> 4;

  __shared__ __align__(16) __bf16 pbuf[4][16 * 32];
  __shared__ __align__(16) float cbuf[4][16];
  __bf16* myp = pbuf[wave];
  float* myc = cbuf[wave];

  const size_t qoff = ((size_t)b * ntok + m0) * 384 + h * 64;
  const v16bf qb0 = load_frag(Q + qoff, 384);       // Q as B-fragment (Q^T)
  const v16bf qb1 = load_frag(Q + qoff + 32, 384);

  v8f oacc[4] = {};
  float mrun = -3.0e38f;
  float lrun = 0.0f;

  for (int j = 0; j < ntok; j += 32) {
    // S^T tiles: rows = keys, cols = queries. K blocks load as A-fragments.
    v8f s0 = {}, s1 = {};
    const size_t koff = ((size_t)b * ntok + j) * 384 + h * 64;
    {
      v16bf ka0 = load_frag(Kmat + koff, 384);
      v16bf ka1 = load_frag(Kmat + koff + 32, 384);
      s0 = WMMA_BF16(ka0, qb0, s0);
      s0 = WMMA_BF16(ka1, qb1, s0);
    }
    {
      v16bf ka0 = load_frag(Kmat + koff + (size_t)16 * 384, 384);
      v16bf ka1 = load_frag(Kmat + koff + (size_t)16 * 384 + 32, 384);
      s1 = WMMA_BF16(ka0, qb0, s1);
      s1 = WMMA_BF16(ka1, qb1, s1);
    }

    // online softmax for query `cl`: 16 in-lane scores + 16 in partner lane (lane^16)
    float mx = fmaxf(s0[0], s1[0]);
#pragma unroll
    for (int i = 1; i < 8; ++i) mx = fmaxf(mx, fmaxf(s0[i], s1[i]));
    mx = fmaxf(mx, __shfl_xor(mx, 16, 32));
    const float mnew = fmaxf(mrun, mx);
    const float corr = __expf(mrun - mnew);
    mrun = mnew;

    v8bf pk0{}, pk1{};
    float rs = 0.0f;
#pragma unroll
    for (int i = 0; i < 8; ++i) {
      const float p0 = __expf(s0[i] - mnew);
      const float p1 = __expf(s1[i] - mnew);
      rs += p0 + p1;
      pk0[i] = (__bf16)p0;
      pk1[i] = (__bf16)p1;
    }
    rs += __shfl_xor(rs, 16, 32);
    lrun = lrun * corr + rs;

    // P^T (regs) -> P row-major [query][32 keys] in LDS: two contiguous 16B stores.
    // tile0 holds keys (half*8 + i), tile1 holds keys (16 + half*8 + i) of query cl.
    *(v8bf*)(myp + cl * 32 + half * 8) = pk0;
    *(v8bf*)(myp + cl * 32 + 16 + half * 8) = pk1;
    myc[cl] = corr;  // lanes cl and cl+16 write the same value
    asm volatile("s_wait_dscnt 0" ::: "memory");

    // per-row corr in C-layout (row = i + 8*half)
    const float4 c0 = *(const float4*)(myc + half * 8);
    const float4 c1 = *(const float4*)(myc + half * 8 + 4);
    const float cr[8] = {c0.x, c0.y, c0.z, c0.w, c1.x, c1.y, c1.z, c1.w};

    const v16bf pf = load_frag(myp, 32);  // P as A-fragment [16 queries x 32 keys]
#pragma unroll
    for (int t = 0; t < 4; ++t) {
#pragma unroll
      for (int i = 0; i < 8; ++i) oacc[t][i] *= cr[i];
      const __bf16* vb = Vt + ((size_t)b * 384 + h * 64 + t * 16) * ntok + j;
      v16bf bfr = load_frag(vb, ntok);  // contiguous per lane thanks to V^T
      oacc[t] = WMMA_BF16(pf, bfr, oacc[t]);
    }
  }

  // broadcast 1/l to C-layout rows, normalize, store O[b, tok, h*64 + col]
  myc[cl] = 1.0f / lrun;
  asm volatile("s_wait_dscnt 0" ::: "memory");
  const float4 c0 = *(const float4*)(myc + half * 8);
  const float4 c1 = *(const float4*)(myc + half * 8 + 4);
  const float inv[8] = {c0.x, c0.y, c0.z, c0.w, c1.x, c1.y, c1.z, c1.w};
#pragma unroll
  for (int t = 0; t < 4; ++t) {
#pragma unroll
    for (int i = 0; i < 8; ++i) {
      const int row = m0 + i + half * 8;
      O[((size_t)b * ntok + row) * 384 + h * 64 + t * 16 + cl] = (__bf16)(oacc[t][i] * inv[i]);
    }
  }
}

// ---------- output projection ----------
// out[b*2048+t, 0:768] = A0[b*2048+t, 0:384] @ Wo[0:384,:]
//                      + (t<1024 ? A1[b*1024+t, 0:384] @ Wo[768:1152,:] : 0) + bo
// Wot: [768][1536] bf16 (transposed Wo). grid = (16384/128, 768/64), block = 256.
__global__ __launch_bounds__(256, 1)
void out_gemm(const __bf16* __restrict__ A0, const __bf16* __restrict__ A1,
              const __bf16* __restrict__ Wot, const float* __restrict__ bo,
              float* __restrict__ out) {
  const int lane = threadIdx.x & 31;
  const int wave = threadIdx.x >> 5;
  const int m0 = blockIdx.x * 128 + wave * 16;  // row in [0, 8*2048)
  const int n0 = blockIdx.y * 64;
  const int b = m0 >> 11;
  const int t0 = m0 & 2047;

  v8f acc[4] = {};
  const __bf16* Abase = A0 + (size_t)m0 * 384;
  for (int kc = 0; kc < 384; kc += 32) {
    v16bf a = load_frag(Abase + kc, 384);
#pragma unroll
    for (int tt = 0; tt < 4; ++tt) {
      v16bf bfr = load_frag(Wot + (size_t)(n0 + tt * 16) * 1536 + kc, 1536);
      acc[tt] = WMMA_BF16(a, bfr, acc[tt]);
    }
  }
  if (t0 < 1024) {  // wave-uniform: tiles never straddle the 1024 boundary
    const __bf16* A1base = A1 + ((size_t)b * 1024 + t0) * 384;
    for (int kc = 0; kc < 384; kc += 32) {
      v16bf a = load_frag(A1base + kc, 384);
#pragma unroll
      for (int tt = 0; tt < 4; ++tt) {
        v16bf bfr = load_frag(Wot + (size_t)(n0 + tt * 16) * 1536 + 768 + kc, 1536);
        acc[tt] = WMMA_BF16(a, bfr, acc[tt]);
      }
    }
  }

  const int cl = lane & 15;
  const int rbase = m0 + ((lane >> 4) << 3);
#pragma unroll
  for (int tt = 0; tt < 4; ++tt) {
    const int col = n0 + tt * 16 + cl;
    const float bv = bo[col];
#pragma unroll
    for (int i = 0; i < 8; ++i)
      out[(size_t)(rbase + i) * 768 + col] = acc[tt][i] + bv;
  }
}

__global__ __launch_bounds__(256, 1) void mask_kernel(float* __restrict__ m) {
  const int i = blockIdx.x * blockDim.x + threadIdx.x;
  if (i < 2 * 2048) {
    const int level = i >> 11;
    const int t = i & 2047;
    m[i] = (level == 0 || t < 1024) ? 1.0f : 0.0f;
  }
}

// ---------- launch ----------
extern "C" void kernel_launch(void* const* d_in, const int* in_sizes, int n_in,
                              void* d_out, int out_size, void* d_ws, size_t ws_size,
                              hipStream_t stream) {
  (void)in_sizes; (void)n_in; (void)out_size; (void)ws_size;
  const float* inp0 = (const float*)d_in[0];
  const float* inp1 = (const float*)d_in[1];
  const float* amask = (const float*)d_in[2];
  const float* Wq0 = (const float*)d_in[3];  const float* bq0 = (const float*)d_in[4];
  const float* Wk0 = (const float*)d_in[5];  const float* bk0 = (const float*)d_in[6];
  const float* Wv0 = (const float*)d_in[7];  const float* bv0 = (const float*)d_in[8];
  const float* Wq1 = (const float*)d_in[9];  const float* bq1 = (const float*)d_in[10];
  const float* Wk1 = (const float*)d_in[11]; const float* bk1 = (const float*)d_in[12];
  const float* Wv1 = (const float*)d_in[13]; const float* bv1 = (const float*)d_in[14];
  const float* Wo  = (const float*)d_in[15]; const float* bo  = (const float*)d_in[16];
  float* out = (float*)d_out;

  // workspace layout (~139 MB total)
  char* ws = (char*)d_ws;
  size_t off = 0;
  auto alloc = [&](size_t bytes) -> char* {
    char* p = ws + off;
    off = (off + bytes + 255) & ~(size_t)255;
    return p;
  };
  __bf16* inp0b = (__bf16*)alloc(16384ull * 768 * 2);
  __bf16* inp1b = (__bf16*)alloc(8192ull * 1536 * 2);
  __bf16* wq0t = (__bf16*)alloc(768ull * 768 * 2);
  __bf16* wk0t = (__bf16*)alloc(768ull * 768 * 2);
  __bf16* wv0t = (__bf16*)alloc(768ull * 768 * 2);
  __bf16* wq1t = (__bf16*)alloc(768ull * 1536 * 2);
  __bf16* wk1t = (__bf16*)alloc(768ull * 1536 * 2);
  __bf16* wv1t = (__bf16*)alloc(768ull * 1536 * 2);
  __bf16* wot  = (__bf16*)alloc(768ull * 1536 * 2);
  __bf16* q0  = (__bf16*)alloc(16384ull * 384 * 2);
  __bf16* k0  = (__bf16*)alloc(16384ull * 384 * 2);
  __bf16* v0t = (__bf16*)alloc(16384ull * 384 * 2);
  __bf16* q1  = (__bf16*)alloc(8192ull * 384 * 2);
  __bf16* k1  = (__bf16*)alloc(8192ull * 384 * 2);
  __bf16* v1t = (__bf16*)alloc(8192ull * 384 * 2);
  __bf16* a0  = (__bf16*)alloc(16384ull * 384 * 2);
  __bf16* a1  = (__bf16*)alloc(8192ull * 384 * 2);

  const int TB = 256;
  auto blocks = [](long long n, int tb) { return (unsigned)((n + tb - 1) / tb); };

  // 1) f32 -> bf16 conversions / weight transposes
  cvt_kernel<<<blocks(16384ll * 768, TB), TB, 0, stream>>>(inp0, inp0b, 16384ll * 768);
  cvt_kernel<<<blocks(8192ll * 1536, TB), TB, 0, stream>>>(inp1, inp1b, 8192ll * 1536);
  cvt_transpose_kernel<<<blocks(768ll * 768, TB), TB, 0, stream>>>(Wq0, wq0t, 768, 768);
  cvt_transpose_kernel<<<blocks(768ll * 768, TB), TB, 0, stream>>>(Wk0, wk0t, 768, 768);
  cvt_transpose_kernel<<<blocks(768ll * 768, TB), TB, 0, stream>>>(Wv0, wv0t, 768, 768);
  cvt_transpose_kernel<<<blocks(1536ll * 768, TB), TB, 0, stream>>>(Wq1, wq1t, 1536, 768);
  cvt_transpose_kernel<<<blocks(1536ll * 768, TB), TB, 0, stream>>>(Wk1, wk1t, 1536, 768);
  cvt_transpose_kernel<<<blocks(1536ll * 768, TB), TB, 0, stream>>>(Wv1, wv1t, 1536, 768);
  cvt_transpose_kernel<<<blocks(1536ll * 768, TB), TB, 0, stream>>>(Wo, wot, 1536, 768);

  // 2) projections (only the 6 used heads per level; q fuses scale+mask; v transposed)
  dim3 gp0(16384 / 128, 6), gp1(8192 / 128, 6);
  proj_gemm<1><<<gp0, 256, 0, stream>>>(inp0b, wq0t, bq0, amask, q0, 768, 2048, 0);
  proj_gemm<0><<<gp0, 256, 0, stream>>>(inp0b, wk0t, bk0, nullptr, k0, 768, 2048, 0);
  proj_gemm<2><<<gp0, 256, 0, stream>>>(inp0b, wv0t, bv0, nullptr, v0t, 768, 2048, 0);
  proj_gemm<1><<<gp1, 256, 0, stream>>>(inp1b, wq1t, bq1, amask, q1, 1536, 1024, 384);
  proj_gemm<0><<<gp1, 256, 0, stream>>>(inp1b, wk1t, bk1, nullptr, k1, 1536, 1024, 384);
  proj_gemm<2><<<gp1, 256, 0, stream>>>(inp1b, wv1t, bv1, nullptr, v1t, 1536, 1024, 384);

  // 3) flash attention per level
  attn_kernel<<<dim3(2048 / 64, 6, 8), 128, 0, stream>>>(q0, k0, v0t, a0, 2048);
  attn_kernel<<<dim3(1024 / 64, 6, 8), 128, 0, stream>>>(q1, k1, v1t, a1, 1024);

  // 4) output projection (skips the structurally-zero halves of x) + mask tail
  out_gemm<<<dim3(16384 / 128, 768 / 64), 256, 0, stream>>>(a0, a1, wot, bo, out);
  mask_kernel<<<dim3(16), 256, 0, stream>>>(out + 8ll * 2048 * 768);
}
